// MultiHeadAttention_53051436040239
// MI455X (gfx1250) — matrix operand order repaired and checked by reading, not compiled, not run on hardware
//
#include <hip/hip_runtime.h>
#include <hip/hip_bf16.h>

typedef __attribute__((ext_vector_type(16))) __bf16 v16bf;
typedef __attribute__((ext_vector_type(8)))  float  v8f;

union BF16Frag { v16bf v; uint4 q[2]; };

#define B_ 4
#define C_ 512
#define T_ 2048
#define H_ 8
#define D_ 64

static __device__ __forceinline__ __bf16 f2bf(float f) { return (__bf16)f; }

// ---------------------------------------------------------------------------
// 1a. Convert the four 512x512 weight matrices fp32 -> bf16 (row-major [o][c])
// ---------------------------------------------------------------------------
__global__ void k_prep_weights(const float* __restrict__ Wq, const float* __restrict__ Wk,
                               const float* __restrict__ Wv, const float* __restrict__ Wo,
                               __bf16* __restrict__ dst) {
  int idx = blockIdx.x * 256 + threadIdx.x;          // 4 * 512*512 total
  const float* srcs[4] = {Wq, Wk, Wv, Wo};
  int m = idx >> 18;                                  // 262144 elems per matrix
  int r = idx & 262143;
  dst[idx] = f2bf(srcs[m][r]);
}

// ---------------------------------------------------------------------------
// 1b. LDS-tiled transpose+convert: x,c [b][c][t] fp32 -> xT,cT [b][t][c] bf16
// ---------------------------------------------------------------------------
__global__ void k_transpose_bf16(const float* __restrict__ x, const float* __restrict__ c,
                                 __bf16* __restrict__ xT, __bf16* __restrict__ cT) {
  __shared__ __bf16 tile[16][17];
  int z = blockIdx.z;                 // 0..7 : (x batches 0..3, c batches 0..3)
  int b = z & 3;
  const float* src = (z < 4) ? x : c;
  __bf16* dst = (z < 4) ? xT : cT;
  int t0 = blockIdx.x * 16, c0 = blockIdx.y * 16;
  int tx = threadIdx.x, ty = threadIdx.y;
  tile[ty][tx] = f2bf(src[((size_t)b * C_ + (c0 + ty)) * T_ + (t0 + tx)]);
  __syncthreads();
  dst[((size_t)b * T_ + (t0 + ty)) * C_ + (c0 + tx)] = tile[tx][ty];
}

// ---------------------------------------------------------------------------
// GEMM tile: out[o0..o0+15][t0..t0+63] = W[o][:] * srcT[t][:] + bias[o]
//   A = W rows (M=o, K=c), B = srcT rows (N=t, K=c contiguous per lane).
// ---------------------------------------------------------------------------
template<bool OUT_F32>
__device__ __forceinline__ void proj_tile(const __bf16* __restrict__ W,
                                          const __bf16* __restrict__ srcT,
                                          const float* __restrict__ bias,
                                          void* __restrict__ dst,
                                          int o0, int t0) {
  int lane = threadIdx.x & 31;
  int n = lane & 15;
  int half = lane >> 4;
  v8f acc[4];
#pragma unroll
  for (int j = 0; j < 4; ++j) acc[j] = (v8f)0.0f;
  const __bf16* wrow = W + (size_t)(o0 + n) * C_;     // A: row M = lane&15
  for (int kc = 0; kc < C_; kc += 32) {
    BF16Frag a;
    a.q[0] = *(const uint4*)(wrow + kc + half * 16);       // K = kc + half*16 + 0..7
    a.q[1] = *(const uint4*)(wrow + kc + half * 16 + 8);   // K = kc + half*16 + 8..15
#pragma unroll
    for (int j = 0; j < 4; ++j) {
      const __bf16* brow = srcT + (size_t)(t0 + j * 16 + n) * C_ + kc + half * 16;
      BF16Frag bb;
      bb.q[0] = *(const uint4*)(brow);
      bb.q[1] = *(const uint4*)(brow + 8);
      acc[j] = __builtin_amdgcn_wmma_f32_16x16x32_bf16(false, a.v, false, bb.v,
                                                       (short)0, acc[j], false, false);
    }
  }
#pragma unroll
  for (int j = 0; j < 4; ++j) {
#pragma unroll
    for (int r = 0; r < 8; ++r) {
      int o = o0 + r + 8 * half;
      int t = t0 + j * 16 + n;
      float v = acc[j][r] + bias[o];
      if (OUT_F32) ((float*)dst)[(size_t)o * T_ + t] = v;
      else        ((__bf16*)dst)[(size_t)o * T_ + t] = f2bf(v);
    }
  }
}

// ---------------------------------------------------------------------------
// 2. Q/K/V projections (bf16 out, [b][c][t])
// ---------------------------------------------------------------------------
__global__ void __launch_bounds__(128)
k_proj_qkv(const __bf16* __restrict__ xT, const __bf16* __restrict__ cT,
           const __bf16* __restrict__ Wall,
           const float* __restrict__ bq, const float* __restrict__ bk, const float* __restrict__ bv,
           __bf16* __restrict__ Qp, __bf16* __restrict__ Kp, __bf16* __restrict__ Vp) {
  int wv = threadIdx.x >> 5;
  int o0 = blockIdx.x * 16;
  int t0 = blockIdx.y * 256 + wv * 64;
  int proj = blockIdx.z >> 2;      // 0:Q 1:K 2:V
  int b = blockIdx.z & 3;
  const __bf16* src = (proj == 0 ? xT : cT) + (size_t)b * T_ * C_;
  const __bf16* W = Wall + (size_t)proj * C_ * C_;
  const float* bias = (proj == 0) ? bq : (proj == 1 ? bk : bv);
  __bf16* dst = ((proj == 0) ? Qp : (proj == 1 ? Kp : Vp)) + (size_t)b * C_ * T_;
  proj_tile<false>(W, src, bias, dst, o0, t0);
}

// ---------------------------------------------------------------------------
// 3. RoPE: Qp/Kp [b][c][t] -> Qr/Kr [b][h][t][64] (rotary applied)
// ---------------------------------------------------------------------------
__global__ void k_rope(const __bf16* __restrict__ Qp, const __bf16* __restrict__ Kp,
                       __bf16* __restrict__ Qr, __bf16* __restrict__ Kr) {
  const __bf16* src = (blockIdx.z == 0) ? Qp : Kp;
  __bf16* dst = (blockIdx.z == 0) ? Qr : Kr;
  int idx = blockIdx.x * 256 + threadIdx.x;     // B*H*T*32 = 2^21
  int j = idx & 31;
  int t = (idx >> 5) & (T_ - 1);
  int bh = idx >> 16;
  int b = bh >> 3, h = bh & 7;
  float x1 = (float)src[((size_t)b * C_ + h * D_ + j) * T_ + t];
  float x2 = (float)src[((size_t)b * C_ + h * D_ + 32 + j) * T_ + t];
  // inv_freq[j] = 10000^(-2j/64) = exp(-j * ln(10000)/32)
  float f = (float)t * __expf(-(float)j * 0.28782313662425572f);
  float sn, cs;
  __sincosf(f, &sn, &cs);
  __bf16* o = dst + ((size_t)bh * T_ + t) * D_;
  o[j]      = f2bf(x1 * cs - x2 * sn);
  o[32 + j] = f2bf(x2 * cs + x1 * sn);
}

// ---------------------------------------------------------------------------
// 4. Flash attention with async-LDS double-buffered K/V staging.
//    One wave = 16 query rows; 8 waves per WG share each K/V tile.
//    Qr,Kr: [b][h][t][64] bf16 ; V = Vp [b][c][t] bf16
//    Output attnT: [b][t][512] bf16
// ---------------------------------------------------------------------------
#define KROW 72   // padded K-tile row: 64 bf16 data + 8 pad (144B, stride-36 banks)
#define VROW 40   // padded V-tile row: 32 bf16 data + 8 pad (80B, stride-20 banks)

__global__ void __launch_bounds__(256)
k_attn(const __bf16* __restrict__ Qr, const __bf16* __restrict__ Kr,
       const __bf16* __restrict__ Vp, __bf16* __restrict__ attnT) {
  __shared__ __bf16 kbuf[2][32][KROW];   // [buf][s within tile][d]
  __shared__ __bf16 vbuf[2][64][VROW];   // [buf][d][s within tile]
  __shared__ __bf16 lds_p[8][16][40];    // per-wave 16x32 P tile, padded rows

  int tid = threadIdx.x;
  int lane = tid & 31;
  int wv = tid >> 5;
  int n = lane & 15, half = lane >> 4;
  int bh = blockIdx.y;
  int b = bh >> 3, h = bh & 7;
  int t0 = blockIdx.x * 128 + wv * 16;

  const __bf16* qbase = Qr + (size_t)bh * T_ * D_;
  const char* kg = (const char*)(Kr + (size_t)bh * T_ * D_);          // row = 128B
  const char* vg = (const char*)(Vp + ((size_t)b * C_ + h * D_) * T_); // row = T_*2 B

  // Cooperative async fill of one K tile (32x64) + one V tile (64x32) into LDS.
  auto issue_fill = [&](int bufi, int s0) {
    {   // K slice: 32 rows x 8 chunks of 16B
      int row = tid >> 3, ch = tid & 7;
      unsigned ldsa = (unsigned)(uintptr_t)&kbuf[bufi][row][ch * 8];
      unsigned long long g = (unsigned long long)(uintptr_t)(kg + (size_t)(s0 + row) * 128 + ch * 16);
      asm volatile("global_load_async_to_lds_b128 %0, %1, off"
                   :: "v"(ldsa), "v"(g) : "memory");
    }
    {   // V slice: 64 rows x 4 chunks of 16B
      int row = tid >> 2, seg = tid & 3;
      unsigned ldsa = (unsigned)(uintptr_t)&vbuf[bufi][row][seg * 8];
      unsigned long long g = (unsigned long long)(uintptr_t)(vg + (size_t)row * (T_ * 2) + (size_t)s0 * 2 + seg * 16);
      asm volatile("global_load_async_to_lds_b128 %0, %1, off"
                   :: "v"(ldsa), "v"(g) : "memory");
    }
  };

  // Q A-fragments for the two 32-wide K(d) chunks (register resident)
  BF16Frag aq[2];
  {
    const __bf16* qrow = qbase + (size_t)(t0 + n) * D_;
#pragma unroll
    for (int kc = 0; kc < 2; ++kc) {
      aq[kc].q[0] = *(const uint4*)(qrow + kc * 32 + half * 16);
      aq[kc].q[1] = *(const uint4*)(qrow + kc * 32 + half * 16 + 8);
    }
  }

  v8f acc[4];
#pragma unroll
  for (int j = 0; j < 4; ++j) acc[j] = (v8f)0.0f;
  float rm[8], rl[8];
#pragma unroll
  for (int r = 0; r < 8; ++r) { rm[r] = -1e30f; rl[r] = 0.0f; }

  constexpr int NIT = T_ / 32;   // 64 tiles of 32 keys
  issue_fill(0, 0);              // pipeline prologue: 2 buffers in flight
  issue_fill(1, 32);             // per-wave ASYNCcnt = 4

  for (int it = 0; it < NIT; ++it) {
    int s0 = it * 32;
    int cur = it & 1;
    // oldest 2 async ops (= current buffer) must be complete; keep next fill in flight
    if (it + 1 < NIT) asm volatile("s_wait_asynccnt 0x2" ::: "memory");
    else              asm volatile("s_wait_asynccnt 0x0" ::: "memory");
    __syncthreads();   // every wave's slice of this buffer is in LDS

    // --- scores: two 16-column tiles, K fragments from LDS ---
    v8f sc[2];
#pragma unroll
    for (int tile = 0; tile < 2; ++tile) {
      const uint4* kr0 = (const uint4*)&kbuf[cur][tile * 16 + n][half * 16];
      const uint4* kr1 = (const uint4*)&kbuf[cur][tile * 16 + n][32 + half * 16];
      BF16Frag bk0, bk1;
      bk0.q[0] = kr0[0]; bk0.q[1] = kr0[1];
      bk1.q[0] = kr1[0]; bk1.q[1] = kr1[1];
      v8f cc = (v8f)0.0f;
      cc = __builtin_amdgcn_wmma_f32_16x16x32_bf16(false, aq[0].v, false, bk0.v, (short)0, cc, false, false);
      cc = __builtin_amdgcn_wmma_f32_16x16x32_bf16(false, aq[1].v, false, bk1.v, (short)0, cc, false, false);
      sc[tile] = cc;
    }
    // --- online softmax; row M = r + 8*half lives across one 16-lane group ---
    float alpha[8];
#pragma unroll
    for (int r = 0; r < 8; ++r) {
      float x0 = sc[0][r] * 0.125f;          // 1/sqrt(64)
      float x1 = sc[1][r] * 0.125f;
      float mx = fmaxf(x0, x1);
#pragma unroll
      for (int off = 8; off >= 1; off >>= 1) mx = fmaxf(mx, __shfl_xor(mx, off, 16));
      float mn = fmaxf(rm[r], mx);
      float p0 = __expf(x0 - mn);
      float p1 = __expf(x1 - mn);
      float rs = p0 + p1;
#pragma unroll
      for (int off = 8; off >= 1; off >>= 1) rs += __shfl_xor(rs, off, 16);
      alpha[r] = __expf(rm[r] - mn);
      rl[r] = rl[r] * alpha[r] + rs;
      rm[r] = mn;
      sc[0][r] = p0;
      sc[1][r] = p1;
    }
#pragma unroll
    for (int j = 0; j < 4; ++j)
#pragma unroll
      for (int r = 0; r < 8; ++r)
        acc[j][r] *= alpha[r];
    // --- P (C layout) -> LDS -> A-layout fragment (wave-local, DS in-order) ---
#pragma unroll
    for (int tile = 0; tile < 2; ++tile)
#pragma unroll
      for (int r = 0; r < 8; ++r)
        lds_p[wv][r + 8 * half][tile * 16 + n] = f2bf(sc[tile][r]);
    asm volatile("s_wait_dscnt 0x0" ::: "memory");
    BF16Frag pa;
    pa.q[0] = *(const uint4*)&lds_p[wv][n][half * 16];
    pa.q[1] = *(const uint4*)&lds_p[wv][n][half * 16 + 8];
    // --- P @ V : V fragments from LDS (rows contiguous along s) ---
#pragma unroll
    for (int j = 0; j < 4; ++j) {
      const uint4* vr = (const uint4*)&vbuf[cur][j * 16 + n][half * 16];
      BF16Frag bv;
      bv.q[0] = vr[0]; bv.q[1] = vr[1];
      acc[j] = __builtin_amdgcn_wmma_f32_16x16x32_bf16(false, pa.v, false, bv.v,
                                                       (short)0, acc[j], false, false);
    }

    __syncthreads();   // all waves done reading buffer `cur` before refilling it
    if (it + 2 < NIT) issue_fill(cur, s0 + 64);
  }

  // epilogue: normalize, store [b][t][h*64+d]
#pragma unroll
  for (int j = 0; j < 4; ++j) {
#pragma unroll
    for (int r = 0; r < 8; ++r) {
      int t = t0 + r + 8 * half;
      int cch = h * D_ + j * 16 + n;
      attnT[((size_t)b * T_ + t) * C_ + cch] = f2bf(acc[j][r] / rl[r]);
    }
  }
}

// ---------------------------------------------------------------------------
// 5. Output projection: Wo * attn + bo  (fp32 out, [b][c][t])
// ---------------------------------------------------------------------------
__global__ void __launch_bounds__(128)
k_proj_out(const __bf16* __restrict__ attnT, const __bf16* __restrict__ Wob,
           const float* __restrict__ bo, float* __restrict__ out) {
  int wv = threadIdx.x >> 5;
  int o0 = blockIdx.x * 16;
  int t0 = blockIdx.y * 256 + wv * 64;
  int b = blockIdx.z;
  proj_tile<true>(Wob, attnT + (size_t)b * T_ * C_, bo,
                  out + (size_t)b * C_ * T_, o0, t0);
}

// ---------------------------------------------------------------------------
extern "C" void kernel_launch(void* const* d_in, const int* in_sizes, int n_in,
                              void* d_out, int out_size, void* d_ws, size_t ws_size,
                              hipStream_t stream) {
  (void)in_sizes; (void)n_in; (void)out_size; (void)ws_size;
  const float* x  = (const float*)d_in[0];
  const float* c  = (const float*)d_in[1];
  const float* Wq = (const float*)d_in[2];
  const float* bq = (const float*)d_in[3];
  const float* Wk = (const float*)d_in[4];
  const float* bk = (const float*)d_in[5];
  const float* Wv = (const float*)d_in[6];
  const float* bv = (const float*)d_in[7];
  const float* Wo = (const float*)d_in[8];
  const float* bo = (const float*)d_in[9];
  float* out = (float*)d_out;

  char* ws = (char*)d_ws;
  size_t off = 0;
  auto alloc = [&](size_t bytes) -> char* {
    char* p = ws + off;
    off += (bytes + 255) & ~(size_t)255;
    return p;
  };
  __bf16* xT    = (__bf16*)alloc((size_t)B_ * T_ * C_ * 2);
  __bf16* cT    = (__bf16*)alloc((size_t)B_ * T_ * C_ * 2);
  __bf16* Wall  = (__bf16*)alloc((size_t)4 * C_ * C_ * 2);
  __bf16* Qp    = (__bf16*)alloc((size_t)B_ * C_ * T_ * 2);
  __bf16* Kp    = (__bf16*)alloc((size_t)B_ * C_ * T_ * 2);
  __bf16* Vp    = (__bf16*)alloc((size_t)B_ * C_ * T_ * 2);
  __bf16* Qr    = (__bf16*)alloc((size_t)B_ * H_ * T_ * D_ * 2);
  __bf16* Kr    = (__bf16*)alloc((size_t)B_ * H_ * T_ * D_ * 2);
  __bf16* attnT = (__bf16*)alloc((size_t)B_ * T_ * C_ * 2);

  k_prep_weights<<<dim3(4096), dim3(256), 0, stream>>>(Wq, Wk, Wv, Wo, Wall);
  k_transpose_bf16<<<dim3(T_ / 16, C_ / 16, 2 * B_), dim3(16, 16), 0, stream>>>(x, c, xT, cT);
  k_proj_qkv<<<dim3(C_ / 16, T_ / 256, 3 * B_), dim3(128), 0, stream>>>(
      xT, cT, Wall, bq, bk, bv, Qp, Kp, Vp);
  k_rope<<<dim3((B_ * H_ * T_ * 32) / 256, 1, 2), dim3(256), 0, stream>>>(Qp, Kp, Qr, Kr);
  k_attn<<<dim3(T_ / 128, B_ * H_), dim3(256), 0, stream>>>(Qr, Kr, Vp, attnT);
  k_proj_out<<<dim3(C_ / 16, T_ / 256, B_), dim3(128), 0, stream>>>(
      attnT, Wall + (size_t)3 * C_ * C_, bo, out);
}
